// TransformerDecoderLayer_equi_v4_21603685499629
// MI455X (gfx1250) — matrix operand order repaired and checked by reading, not compiled, not run on hardware
//
#include <hip/hip_runtime.h>

#define P_ 20000
#define M_ 20000
#define N_ 16
#define D_ 128
#define H_ 8
#define DFF_ 512

typedef __attribute__((ext_vector_type(16))) __bf16 v16bf;
typedef __attribute__((ext_vector_type(8)))  float  v8f;

#if __has_builtin(__builtin_amdgcn_tensor_load_to_lds) && \
    __has_builtin(__builtin_amdgcn_s_wait_tensorcnt)
#define USE_TDM 1
typedef __attribute__((ext_vector_type(4))) unsigned int u32x4_t;
typedef __attribute__((ext_vector_type(8))) int i32x8_t;
typedef __attribute__((ext_vector_type(4))) int i32x4_t;
#else
#define USE_TDM 0
#endif

__device__ inline v8f v8f_zero() {
    v8f z = {0.f, 0.f, 0.f, 0.f, 0.f, 0.f, 0.f, 0.f};
    return z;
}

__device__ inline v8f wmma_bf16(v16bf a, v16bf b, v8f c) {
    return __builtin_amdgcn_wmma_f32_16x16x32_bf16(false, a, false, b, (short)0, c, false, false);
}

// A fragment (16x32 bf16, M x K) from an LDS tile stored row-major [16][ld].
// Lane L: row m = L&15; half = L>>4 selects K offset {0,8}; VGPRs 0-3 hold K base,
// VGPRs 4-7 hold K base+16 (CDNA5 ISA 7.12.2). Two contiguous 8-element runs
// per lane -> compiler emits ds_load_b128 pairs.
__device__ inline v16bf load_a_frag(const __bf16* __restrict__ A, int ld, int k0, int lane) {
    int m = lane & 15, half = lane >> 4;
    const __bf16* row = A + m * ld + k0 + half * 8;
    v16bf a;
#pragma unroll
    for (int idx = 0; idx < 16; ++idx) {
        int k = (idx & 7) + ((idx >> 3) << 4);
        a[idx] = row[k];
    }
    return a;
}

// B fragment from a pre-packed fragment-major bf16 weight image: each (nt,kt,lane)
// owns 16 contiguous bf16 (32 B, aligned) -> a single wide global load from L2.
__device__ inline v16bf load_b_packed(const __bf16* __restrict__ Wp, int KT, int nt,
                                      int kt, int lane) {
    return *(const v16bf*)(Wp + (((size_t)nt * KT + kt) * 32 + lane) * 16);
}

// ---------------------------------------------------------------------------
// One-time weight packer: f32 (K x ld) -> fragment-major bf16 image matching the
// CDNA5 B-matrix VGPR layout (lane n = L&15 col, half = L>>4 selects K offset {0,8}; elems 8-15 at K+16).
// ---------------------------------------------------------------------------
__global__ __launch_bounds__(256) void k_pack_w(const float* __restrict__ W,
                                                __bf16* __restrict__ out,
                                                int K, int ld, int ncols, int nvalid) {
    int w = blockIdx.x * 8 + (threadIdx.x >> 5);
    int lane = threadIdx.x & 31;
    int KT = K >> 5, NT = ncols >> 4;
    if (w >= KT * NT) return;
    int nt = w / KT, kt = w - nt * KT;
    int n = lane & 15, half = lane >> 4;
    __bf16* dst = out + ((size_t)w * 32 + lane) * 16;
    int col = nt * 16 + n;
    bool ok = col < nvalid;
#pragma unroll
    for (int idx = 0; idx < 16; ++idx) {
        int k = kt * 32 + half * 8 + (idx & 7) + ((idx >> 3) << 4);
        dst[idx] = (__bf16)(ok ? W[(size_t)k * ld + col] : 0.f);
    }
}

// ---------------------------------------------------------------------------
// VN linear: Y[p,o,i] = sum_c X[p,c,i] * W[c,o]   (Cin = Cout = 128)
// One block = 16 points; 8 waves each own a 16-col tile; 3 accumulators (x,y,z).
// ---------------------------------------------------------------------------
__global__ __launch_bounds__(256) void k_vn_gemm(const float* __restrict__ X,
                                                 const __bf16* __restrict__ Wp,
                                                 float* __restrict__ Y) {
    __shared__ __bf16 sA[3 * 16 * 128];
    int tid = threadIdx.x;
    int p0  = blockIdx.x * 16;
    for (int e = tid; e < 16 * 128 * 3; e += 256) {
        int i = e % 3, c = (e / 3) & 127, m = e / 384;
        sA[(i * 16 + m) * 128 + c] = (__bf16)X[((size_t)(p0 + m) * 128 + c) * 3 + i];
    }
    __syncthreads();
    int wv = tid >> 5, lane = tid & 31, n0 = wv * 16;
    v8f a0 = v8f_zero(), a1 = v8f_zero(), a2 = v8f_zero();
#pragma unroll
    for (int k0 = 0; k0 < 128; k0 += 32) {
        v16bf bf = load_b_packed(Wp, 4, wv, k0 >> 5, lane);
        a0 = wmma_bf16(load_a_frag(sA + 0 * 2048, 128, k0, lane), bf, a0);
        a1 = wmma_bf16(load_a_frag(sA + 1 * 2048, 128, k0, lane), bf, a1);
        a2 = wmma_bf16(load_a_frag(sA + 2 * 2048, 128, k0, lane), bf, a2);
    }
    int n = lane & 15, mb = (lane >> 4) * 8;
#pragma unroll
    for (int j = 0; j < 8; ++j) {
        size_t o = ((size_t)(p0 + mb + j) * 128 + (n0 + n)) * 3;
        Y[o + 0] = a0[j];
        Y[o + 1] = a1[j];
        Y[o + 2] = a2[j];
    }
}

// ---------------------------------------------------------------------------
// Fused dual VN linear (K and V projections share one pass over memory):
// stage A once, run two packed weights -> two outputs. Saves a 30 MB HBM pass.
// ---------------------------------------------------------------------------
__global__ __launch_bounds__(256) void k_vn_gemm2(const float* __restrict__ X,
                                                  const __bf16* __restrict__ Wp1,
                                                  const __bf16* __restrict__ Wp2,
                                                  float* __restrict__ Y1,
                                                  float* __restrict__ Y2) {
    __shared__ __bf16 sA[3 * 16 * 128];
    int tid = threadIdx.x;
    int p0  = blockIdx.x * 16;
    for (int e = tid; e < 16 * 128 * 3; e += 256) {
        int i = e % 3, c = (e / 3) & 127, m = e / 384;
        sA[(i * 16 + m) * 128 + c] = (__bf16)X[((size_t)(p0 + m) * 128 + c) * 3 + i];
    }
    __syncthreads();
    int wv = tid >> 5, lane = tid & 31, n0 = wv * 16;
    v8f k0a = v8f_zero(), k1a = v8f_zero(), k2a = v8f_zero();
    v8f v0a = v8f_zero(), v1a = v8f_zero(), v2a = v8f_zero();
#pragma unroll
    for (int k0 = 0; k0 < 128; k0 += 32) {
        v16bf fa0 = load_a_frag(sA + 0 * 2048, 128, k0, lane);
        v16bf fa1 = load_a_frag(sA + 1 * 2048, 128, k0, lane);
        v16bf fa2 = load_a_frag(sA + 2 * 2048, 128, k0, lane);
        v16bf b1 = load_b_packed(Wp1, 4, wv, k0 >> 5, lane);
        v16bf b2 = load_b_packed(Wp2, 4, wv, k0 >> 5, lane);
        k0a = wmma_bf16(fa0, b1, k0a);
        k1a = wmma_bf16(fa1, b1, k1a);
        k2a = wmma_bf16(fa2, b1, k2a);
        v0a = wmma_bf16(fa0, b2, v0a);
        v1a = wmma_bf16(fa1, b2, v1a);
        v2a = wmma_bf16(fa2, b2, v2a);
    }
    int n = lane & 15, mb = (lane >> 4) * 8;
#pragma unroll
    for (int j = 0; j < 8; ++j) {
        size_t o = ((size_t)(p0 + mb + j) * 128 + (n0 + n)) * 3;
        Y1[o + 0] = k0a[j];
        Y1[o + 1] = k1a[j];
        Y1[o + 2] = k2a[j];
        Y2[o + 0] = v0a[j];
        Y2[o + 1] = v1a[j];
        Y2[o + 2] = v2a[j];
    }
}

// ---------------------------------------------------------------------------
// Fused neighbor attention: one block per point.
// Wave 0 issues two TDM gather descriptors (8 x 32-bit row indices each) that
// DMA the 16 gathered v rows straight into LDS, overlapped with the k stream.
// ---------------------------------------------------------------------------
__global__ __launch_bounds__(256) void k_attn(const float* __restrict__ q,
                                              const float* __restrict__ k_all,
                                              const float* __restrict__ v_all,
                                              const float* __restrict__ sh,
                                              const float* __restrict__ dist,
                                              const int* __restrict__ idxp,
                                              const __bf16* __restrict__ gw1p,
                                              const float* __restrict__ gb1,
                                              const float* __restrict__ gw2,
                                              const float* __restrict__ gb2,
                                              float* __restrict__ tgt2) {
    __shared__ float  sQ[128 * 3];
    __shared__ float  sSh[16 * 3];
    __shared__ float  sV[16 * 128 * 3];
    __shared__ __bf16 sPos[16 * 128];
    __shared__ float  sDot[16 * 8];
    __shared__ float  sHid[16 * 8];
    __shared__ float  sAtt[16 * 8];

    int tid = threadIdx.x;
    int p   = blockIdx.x;
    int wv  = tid >> 5, lane = tid & 31;

#if USE_TDM
    if (wv == 0) {
        // Tensor DMA gather: D# per CDNA5 ISA ch.8 (gather_mode, 32-bit indices).
        unsigned ldsAddr = (unsigned)(unsigned long long)(const void*)sV;
        unsigned long long gbase = (unsigned long long)v_all;
        unsigned g0w0 = 1u | (1u << 30) | (1u << 31);       // count=1, idx32, gather
        unsigned g0w2 = (unsigned)gbase;
        unsigned g0w3 = (unsigned)((gbase >> 32) & 0x01FFFFFFu) | (2u << 30); // type=2
        u32x4_t g0 = {g0w0, ldsAddr, g0w2, g0w3};
        i32x8_t g1;
        g1[0] = (2 << 16);                 // data_size = 4B
        g1[1] = (int)(384u << 16);         // tensor_dim0 lo16 (=384)
        g1[2] = (int)((unsigned)M_ << 16); // tensor_dim0 hi16=0 | tensor_dim1 lo16
        g1[3] = (int)(384u << 16);         // tensor_dim1 hi16=0 | tile_dim0 = 384
        g1[4] = 8;                         // tile_dim1 = 8 gather indices
        g1[5] = 384;                       // tensor_dim0_stride (48b) lo32
        g1[6] = 0;
        g1[7] = 0;
        i32x8_t g4 = {0, 0, 0, 0, 0, 0, 0, 0};  // reserved VADDR4 group (NULL)
        const int* ip = idxp + (size_t)p * 16;
        int i0 = __builtin_amdgcn_readfirstlane(ip[0]);
        int i1 = __builtin_amdgcn_readfirstlane(ip[1]);
        int i2 = __builtin_amdgcn_readfirstlane(ip[2]);
        int i3 = __builtin_amdgcn_readfirstlane(ip[3]);
        int i4 = __builtin_amdgcn_readfirstlane(ip[4]);
        int i5 = __builtin_amdgcn_readfirstlane(ip[5]);
        int i6 = __builtin_amdgcn_readfirstlane(ip[6]);
        int i7 = __builtin_amdgcn_readfirstlane(ip[7]);
        i32x4_t g2 = {i0, i1, i2, i3};
        i32x4_t g3 = {i4, i5, i6, i7};
        __builtin_amdgcn_tensor_load_to_lds(g0, g1, g2, g3, g4, 0);
        int i8  = __builtin_amdgcn_readfirstlane(ip[8]);
        int i9  = __builtin_amdgcn_readfirstlane(ip[9]);
        int i10 = __builtin_amdgcn_readfirstlane(ip[10]);
        int i11 = __builtin_amdgcn_readfirstlane(ip[11]);
        int i12 = __builtin_amdgcn_readfirstlane(ip[12]);
        int i13 = __builtin_amdgcn_readfirstlane(ip[13]);
        int i14 = __builtin_amdgcn_readfirstlane(ip[14]);
        int i15 = __builtin_amdgcn_readfirstlane(ip[15]);
        u32x4_t g0b = {g0w0, ldsAddr + 8u * 1536u, g0w2, g0w3};
        i32x4_t g2b = {i8, i9, i10, i11};
        i32x4_t g3b = {i12, i13, i14, i15};
        __builtin_amdgcn_tensor_load_to_lds(g0b, g1, g2b, g3b, g4, 0);
    }
#endif

    for (int e = tid; e < 384; e += 256) sQ[e] = q[(size_t)p * 384 + e];
    if (tid < 48) sSh[tid] = sh[(size_t)p * 51 + 3 + tid];  // drop l=0 term
    __syncthreads();

    int h4 = lane >> 2;  // head id for this lane's d-range (d = lane*4 + jj)
    for (int nn = wv; nn < 16; nn += 8) {
        int idx = idxp[(size_t)p * 16 + nn];
        if (nn + 8 < 16) {
            int idx2 = idxp[(size_t)p * 16 + nn + 8];
            __builtin_prefetch(k_all + (size_t)idx2 * 384 + lane * 12, 0, 0);
        }
        const float4* kp = (const float4*)(k_all + (size_t)idx * 384) + lane * 3;
        float4 k0 = kp[0], k1 = kp[1], k2 = kp[2];
#if !USE_TDM
        const float4* vp = (const float4*)(v_all + (size_t)idx * 384) + lane * 3;
        float4 v0 = vp[0], v1 = vp[1], v2 = vp[2];
        float4* svp = (float4*)(sV + nn * 384) + lane * 3;
        svp[0] = v0; svp[1] = v1; svp[2] = v2;
#endif
        float kk[12] = {k0.x, k0.y, k0.z, k0.w, k1.x, k1.y, k1.z, k1.w,
                        k2.x, k2.y, k2.z, k2.w};
        float shx = sSh[nn * 3 + 0], shy = sSh[nn * 3 + 1], shz = sSh[nn * 3 + 2];
        float dotp = 0.f;
#pragma unroll
        for (int jj = 0; jj < 4; ++jj) {
            int d = lane * 4 + jj;
            float kx = kk[jj * 3 + 0], ky = kk[jj * 3 + 1], kz = kk[jj * 3 + 2];
            float qx = sQ[d * 3 + 0], qy = sQ[d * 3 + 1], qz = sQ[d * 3 + 2];
            dotp += kx * qx + ky * qy + kz * qz;
            float pos = (kx - qx) * shx + (ky - qy) * shy + (kz - qz) * shz;
            sPos[nn * 128 + d] = (__bf16)pos;
        }
        dotp += __shfl_xor(dotp, 1);
        dotp += __shfl_xor(dotp, 2);
        if ((lane & 3) == 0) sDot[nn * 8 + h4] = dotp;
    }
    __syncthreads();

    // pos MLP layer 1 via WMMA: (16 x 128) @ (128 x 8, zero-padded to 16)
    if (wv == 0) {
        v8f acc = v8f_zero();
#pragma unroll
        for (int k0 = 0; k0 < 128; k0 += 32) {
            acc = wmma_bf16(load_a_frag(sPos, 128, k0, lane),
                            load_b_packed(gw1p, 4, 0, k0 >> 5, lane), acc);
        }
        int n = lane & 15, mb = (lane >> 4) * 8;
        if (n < 8) {
            float bb = gb1[n];
#pragma unroll
            for (int j = 0; j < 8; ++j)
                sHid[(mb + j) * 8 + n] = fmaxf(acc[j] + bb, 0.f);
        }
    }
    __syncthreads();

    if (tid < 128) {  // layer 2: (16x8)@(8x8), + dot + dist, scale 1/sqrt(DPH)=0.25
        int nn = tid >> 3, h = tid & 7;
        float s = gb2[h];
#pragma unroll
        for (int j = 0; j < 8; ++j) s += sHid[nn * 8 + j] * gw2[j * 8 + h];
        sAtt[nn * 8 + h] =
            (sDot[nn * 8 + h] + s + dist[((size_t)p * 16 + nn) * 8 + h]) * 0.25f;
    }
    __syncthreads();

    if (tid < 8) {  // softmax over 16 neighbors, per head
        int h = tid;
        float mx = -3.4e38f;
#pragma unroll
        for (int nn = 0; nn < 16; ++nn) mx = fmaxf(mx, sAtt[nn * 8 + h]);
        float ex[16], sum = 0.f;
#pragma unroll
        for (int nn = 0; nn < 16; ++nn) {
            ex[nn] = __expf(sAtt[nn * 8 + h] - mx);
            sum += ex[nn];
        }
        float inv = 1.f / sum;
#pragma unroll
        for (int nn = 0; nn < 16; ++nn) sAtt[nn * 8 + h] = ex[nn] * inv;
    }
#if USE_TDM
    if (wv == 0) __builtin_amdgcn_s_wait_tensorcnt(0);  // sV DMA complete
#endif
    __syncthreads();

    for (int e = tid; e < 384; e += 256) {
        int d = e / 3, i = e - d * 3, h = d >> 4;
        float acc = 0.f;
#pragma unroll
        for (int nn = 0; nn < 16; ++nn)
            acc += sAtt[nn * 8 + h] * sV[(nn * 128 + d) * 3 + i];
        tgt2[(size_t)p * 384 + e] = acc;
    }
}

// ---------------------------------------------------------------------------
// VN layernorm over a 16-point block held in LDS (sY: [16][128][3]).
// ---------------------------------------------------------------------------
__device__ inline void block_vn_layernorm(const float* __restrict__ gamma,
                                          const float* __restrict__ beta,
                                          float* sY, float* sN, float* sMu, float* sRv,
                                          float* __restrict__ outp, int p0, int tid) {
    for (int e = tid; e < 16 * 128; e += 256) {
        const float* v = &sY[e * 3];
        sN[e] = sqrtf(v[0] * v[0] + v[1] * v[1] + v[2] * v[2]);
    }
    __syncthreads();
    if (tid < 16) {
        const float* row = &sN[tid * 128];
        float s1 = 0.f, s2 = 0.f;
        for (int d = 0; d < 128; ++d) {
            float n = row[d];
            s1 += n;
            s2 += n * n;
        }
        float mu = s1 * (1.f / 128.f);
        float var = s2 * (1.f / 128.f) - mu * mu;
        sMu[tid] = mu;
        sRv[tid] = rsqrtf(var + 1e-5f);
    }
    __syncthreads();
    for (int e = tid; e < 16 * 128; e += 256) {
        int m = e >> 7, d = e & 127;
        float n = sN[e];
        float nh = (n - sMu[m]) * sRv[m] * gamma[d] + beta[d];
        float sc = nh / (n + 1e-6f);
        size_t o = ((size_t)(p0 + m) * 128 + d) * 3;
        outp[o + 0] = sY[e * 3 + 0] * sc;
        outp[o + 1] = sY[e * 3 + 1] * sc;
        outp[o + 2] = sY[e * 3 + 2] * sc;
    }
}

// ---------------------------------------------------------------------------
// tgt2 @ Wo + residual(x) -> VN layernorm (ln1) -> y1.
// ---------------------------------------------------------------------------
__global__ __launch_bounds__(256) void k_wo_res_ln(const float* __restrict__ t2,
                                                   const __bf16* __restrict__ Wop,
                                                   const float* __restrict__ x,
                                                   const float* __restrict__ g,
                                                   const float* __restrict__ b,
                                                   float* __restrict__ y1) {
    __shared__ __bf16 sA[3 * 16 * 128];
    __shared__ float  sY[16 * 128 * 3];
    __shared__ float  sN[16 * 128];
    __shared__ float  sMu[16], sRv[16];
    int tid = threadIdx.x;
    int p0  = blockIdx.x * 16;
    for (int e = tid; e < 6144; e += 256) {
        int i = e % 3, c = (e / 3) & 127, m = e / 384;
        sA[(i * 16 + m) * 128 + c] = (__bf16)t2[((size_t)(p0 + m) * 128 + c) * 3 + i];
    }
    __syncthreads();
    int wv = tid >> 5, lane = tid & 31, n0 = wv * 16;
    v8f a0 = v8f_zero(), a1 = v8f_zero(), a2 = v8f_zero();
#pragma unroll
    for (int k0 = 0; k0 < 128; k0 += 32) {
        v16bf bf = load_b_packed(Wop, 4, wv, k0 >> 5, lane);
        a0 = wmma_bf16(load_a_frag(sA + 0 * 2048, 128, k0, lane), bf, a0);
        a1 = wmma_bf16(load_a_frag(sA + 1 * 2048, 128, k0, lane), bf, a1);
        a2 = wmma_bf16(load_a_frag(sA + 2 * 2048, 128, k0, lane), bf, a2);
    }
    int n = lane & 15, mb = (lane >> 4) * 8;
#pragma unroll
    for (int j = 0; j < 8; ++j) {
        int m = mb + j;
        size_t o = ((size_t)(p0 + m) * 128 + (n0 + n)) * 3;
        int le = (m * 128 + (n0 + n)) * 3;
        sY[le + 0] = a0[j] + x[o + 0];
        sY[le + 1] = a1[j] + x[o + 1];
        sY[le + 2] = a2[j] + x[o + 2];
    }
    __syncthreads();
    block_vn_layernorm(g, b, sY, sN, sMu, sRv, y1, p0, tid);
}

// ---------------------------------------------------------------------------
// Fused FFN: u=y@ff_w1, t=y@ff_d1 (128->512), vn_relu, h1@ff_w2 (512->128),
// + residual(y1) -> VN layernorm (ln2) -> out. h1 lives entirely in LDS.
// LDS aliasing: [0,12K)=sAy(bf16) then reused as sN/sMu/sRv;
//               [12K,60K)=sH1(bf16) then reused as sY(f32).
// ---------------------------------------------------------------------------
__global__ __launch_bounds__(256) void k_ffn(const float* __restrict__ y1,
                                             const __bf16* __restrict__ w1p,
                                             const __bf16* __restrict__ d1p,
                                             const __bf16* __restrict__ w2p,
                                             const float* __restrict__ g,
                                             const float* __restrict__ b,
                                             float* __restrict__ out) {
    __shared__ __align__(16) unsigned char smem[61440];
    __bf16* sAy = (__bf16*)smem;                 // 3*16*128 bf16 = 12288 B
    __bf16* sH1 = (__bf16*)(smem + 12288);       // 48*512 bf16   = 49152 B
    float*  sY  = (float*)(smem + 12288);        // 16*128*3 f32  = 24576 B (aliases sH1)
    float*  sN  = (float*)smem;                  // 16*128 f32    =  8192 B (aliases sAy)
    float*  sMu = (float*)(smem + 8192);
    float*  sRv = (float*)(smem + 8192 + 64);

    int tid = threadIdx.x;
    int p0  = blockIdx.x * 16;
    for (int e = tid; e < 6144; e += 256) {
        int i = e % 3, c = (e / 3) & 127, m = e / 384;
        sAy[(i * 16 + m) * 128 + c] = (__bf16)y1[((size_t)(p0 + m) * 128 + c) * 3 + i];
    }
    __syncthreads();

    int wv = tid >> 5, lane = tid & 31;
    int n = lane & 15, mb = (lane >> 4) * 8;

    // First GEMM pair + vn_relu; each wave owns 4 of the 32 DFF col-tiles.
    for (int ct = wv; ct < 32; ct += 8) {
        v8f u0 = v8f_zero(), u1 = v8f_zero(), u2 = v8f_zero();
        v8f t0 = v8f_zero(), t1 = v8f_zero(), t2 = v8f_zero();
#pragma unroll
        for (int k0 = 0; k0 < 128; k0 += 32) {
            v16bf fa0 = load_a_frag(sAy + 0 * 2048, 128, k0, lane);
            v16bf fa1 = load_a_frag(sAy + 1 * 2048, 128, k0, lane);
            v16bf fa2 = load_a_frag(sAy + 2 * 2048, 128, k0, lane);
            v16bf bu  = load_b_packed(w1p, 4, ct, k0 >> 5, lane);
            v16bf bt  = load_b_packed(d1p, 4, ct, k0 >> 5, lane);
            u0 = wmma_bf16(fa0, bu, u0);
            u1 = wmma_bf16(fa1, bu, u1);
            u2 = wmma_bf16(fa2, bu, u2);
            t0 = wmma_bf16(fa0, bt, t0);
            t1 = wmma_bf16(fa1, bt, t1);
            t2 = wmma_bf16(fa2, bt, t2);
        }
#pragma unroll
        for (int j = 0; j < 8; ++j) {
            int m = mb + j;
            float ux = u0[j], uy = u1[j], uz = u2[j];
            float tx = t0[j], ty = t1[j], tz = t2[j];
            float dot = ux * tx + uy * ty + uz * tz;
            float ksq = tx * tx + ty * ty + tz * tz;
            if (dot < 0.f) {
                float f = dot / (ksq + 1e-6f);
                ux -= f * tx; uy -= f * ty; uz -= f * tz;
            }
            int c = ct * 16 + n;
            sH1[(0 * 16 + m) * 512 + c] = (__bf16)ux;
            sH1[(1 * 16 + m) * 512 + c] = (__bf16)uy;
            sH1[(2 * 16 + m) * 512 + c] = (__bf16)uz;
        }
    }
    __syncthreads();

    // Second GEMM: h1 (48 x 512) @ ff_w2 (512 x 128); one col-tile per wave.
    {
        int n0 = wv * 16;
        v8f o0 = v8f_zero(), o1 = v8f_zero(), o2 = v8f_zero();
        for (int k0 = 0; k0 < 512; k0 += 32) {
            v16bf bf = load_b_packed(w2p, 16, wv, k0 >> 5, lane);
            o0 = wmma_bf16(load_a_frag(sH1 + 0 * 16 * 512, 512, k0, lane), bf, o0);
            o1 = wmma_bf16(load_a_frag(sH1 + 1 * 16 * 512, 512, k0, lane), bf, o1);
            o2 = wmma_bf16(load_a_frag(sH1 + 2 * 16 * 512, 512, k0, lane), bf, o2);
        }
        __syncthreads();  // everyone done reading sH1 before overwriting as sY
#pragma unroll
        for (int j = 0; j < 8; ++j) {
            int m = mb + j;
            size_t o = ((size_t)(p0 + m) * 128 + (n0 + n)) * 3;
            int le = (m * 128 + (n0 + n)) * 3;
            sY[le + 0] = o0[j] + y1[o + 0];
            sY[le + 1] = o1[j] + y1[o + 1];
            sY[le + 2] = o2[j] + y1[o + 2];
        }
    }
    __syncthreads();
    block_vn_layernorm(g, b, sY, sN, sMu, sRv, out, p0, tid);
}

extern "C" void kernel_launch(void* const* d_in, const int* in_sizes, int n_in,
                              void* d_out, int out_size, void* d_ws, size_t ws_size,
                              hipStream_t stream) {
    const float* tgt  = (const float*)d_in[0];
    const float* mem  = (const float*)d_in[1];
    const float* sh   = (const float*)d_in[2];
    const float* dist = (const float*)d_in[3];
    const float* Wq   = (const float*)d_in[4];
    const float* Wk   = (const float*)d_in[5];
    const float* Wv   = (const float*)d_in[6];
    const float* Wo   = (const float*)d_in[7];
    const float* gw1  = (const float*)d_in[8];
    const float* gb1  = (const float*)d_in[9];
    const float* gw2  = (const float*)d_in[10];
    const float* gb2  = (const float*)d_in[11];
    const float* ln1g = (const float*)d_in[12];
    const float* ln1b = (const float*)d_in[13];
    const float* ln2g = (const float*)d_in[14];
    const float* ln2b = (const float*)d_in[15];
    const float* ffw1 = (const float*)d_in[16];
    const float* ffd1 = (const float*)d_in[17];
    const float* ffw2 = (const float*)d_in[18];
    const int*   idxp = (const int*)d_in[19];

    float* ws   = (float*)d_ws;
    float* q    = ws;                              // (P,128,3)
    float* kall = q + (size_t)P_ * 384;            // (M,128,3)
    float* vall = kall + (size_t)M_ * 384;         // (M,128,3)
    float* t2   = vall + (size_t)M_ * 384;         // (P,128,3)
    float* y1   = q;                               // reuse: q dead after k_attn

    __bf16* wp   = (__bf16*)(t2 + (size_t)P_ * 384);
    __bf16* Wqp  = wp;                 // 128x128 -> 16384
    __bf16* Wkp  = Wqp + 16384;
    __bf16* Wvp  = Wkp + 16384;
    __bf16* Wop  = Wvp + 16384;
    __bf16* gw1p = Wop + 16384;        // 128x16 (8 valid) -> 2048
    __bf16* w1p  = gw1p + 2048;        // 128x512 -> 65536
    __bf16* d1p  = w1p + 65536;
    __bf16* w2p  = d1p + 65536;        // 512x128 -> 65536
    float* outp = (float*)d_out;

    dim3 blk(256);
    // Pack all weights once into WMMA-fragment-major bf16 images.
    k_pack_w<<<4, blk, 0, stream>>>(Wq, Wqp, 128, 128, 128, 128);
    k_pack_w<<<4, blk, 0, stream>>>(Wk, Wkp, 128, 128, 128, 128);
    k_pack_w<<<4, blk, 0, stream>>>(Wv, Wvp, 128, 128, 128, 128);
    k_pack_w<<<4, blk, 0, stream>>>(Wo, Wop, 128, 128, 128, 128);
    k_pack_w<<<1, blk, 0, stream>>>(gw1, gw1p, 128, 8, 16, 8);
    k_pack_w<<<16, blk, 0, stream>>>(ffw1, w1p, 128, 512, 512, 512);
    k_pack_w<<<16, blk, 0, stream>>>(ffd1, d1p, 128, 512, 512, 512);
    k_pack_w<<<16, blk, 0, stream>>>(ffw2, w2p, 512, 128, 128, 128);

    k_vn_gemm<<<P_ / 16, blk, 0, stream>>>(tgt, Wqp, q);
    k_vn_gemm2<<<M_ / 16, blk, 0, stream>>>(mem, Wkp, Wvp, kall, vall);
    k_attn<<<P_, blk, 0, stream>>>(q, kall, vall, sh, dist, idxp,
                                   gw1p, gb1, gw2, gb2, t2);
    k_wo_res_ln<<<P_ / 16, blk, 0, stream>>>(t2, Wop, tgt, ln1g, ln1b, y1);
    k_ffn<<<P_ / 16, blk, 0, stream>>>(y1, w1p, d1p, w2p, ln2g, ln2b, outp);
}
